// QuasicrystalAttention_24730421690772
// MI455X (gfx1250) — compile-verified, matrix operand order checked by reading, not run on hardware
//
#include <hip/hip_runtime.h>

typedef _Float16 v16h __attribute__((ext_vector_type(16)));
typedef _Float16 v8h  __attribute__((ext_vector_type(8)));
typedef float    v8f  __attribute__((ext_vector_type(8)));

#define BB 8
#define LL 1024
#define DD 1280
#define HH 5
#define HD 256
#define N3 3840

__device__ __forceinline__ v8f wmma_f16(v16h a, v16h b, v8f c) {
  return __builtin_amdgcn_wmma_f32_16x16x32_f16(false, a, false, b, (short)0, c,
                                                false, false);
}

// Load a 16x32 f16 fragment (A-layout; B uses the same layout against a
// transposed operand). p points at element (0,0) of the fragment; rows are
// `ld` f16 apart. Lane l -> row m=l&15, K base kb=(l>>4)*8; elems 0-7 at
// K=kb..kb+7, elems 8-15 at K=16+kb..16+kb+7 (two 16B chunks).
__device__ __forceinline__ v16h ld_frag(const _Float16* __restrict__ p, int ld) {
  unsigned lane = threadIdx.x & 31u;
  unsigned m  = lane & 15u;
  unsigned kb = (lane >> 4) << 3;
  const _Float16* q = p + (size_t)m * (size_t)ld + kb;
  union { v16h v; v8h h[2]; } u;
  u.h[0] = *(const v8h*)(q);
  u.h[1] = *(const v8h*)(q + 16);
  return u.v;
}

// ---------------- conversion kernels ----------------

__global__ void cvt_f32_to_f16(const float* __restrict__ in,
                               _Float16* __restrict__ out, int n) {
  int i = blockIdx.x * blockDim.x + threadIdx.x;
  if (i < n) out[i] = (_Float16)in[i];
}

// W[K][N] (f32, row-major) -> Wt[N][K] (f16), LDS-tiled for coalescing.
// Block: 256 threads handling a 32x32 tile. Grid: (K/32, N/32).
__global__ void __launch_bounds__(256) transpose_cvt(
    const float* __restrict__ W, _Float16* __restrict__ Wt, int K, int N) {
  __shared__ float tile[32][33];
  int kb = blockIdx.x * 32;
  int nb = blockIdx.y * 32;
  int tx = (int)(threadIdx.x & 31u);
  int ty = (int)(threadIdx.x >> 5);  // 0..7
#pragma unroll
  for (int i = 0; i < 32; i += 8)
    tile[ty + i][tx] = W[(size_t)(kb + ty + i) * N + (nb + tx)];
  __syncthreads();
#pragma unroll
  for (int i = 0; i < 32; i += 8)
    Wt[(size_t)(nb + ty + i) * K + (kb + tx)] = (_Float16)tile[tx][ty + i];
}

// ---------------- QKV projection GEMM ----------------
// A: x_h [8192][1280] f16 ; Bt: WqkvT [3840][1280] f16
// Epilogue: bias + phase modulation, scatter into q ([B,H,L,hd], pre-scaled by
// hd^-0.5), k ([B,H,L,hd]) and v transposed ([B,H,hd,L]).
__global__ void __launch_bounds__(128) gemm_qkv(
    const _Float16* __restrict__ A, const _Float16* __restrict__ Bt,
    const float* __restrict__ bias, const float* __restrict__ phi,
    _Float16* __restrict__ qh, _Float16* __restrict__ kh,
    _Float16* __restrict__ vt) {
  const int K = DD;
  unsigned wid  = threadIdx.x >> 5;
  unsigned lane = threadIdx.x & 31u;
  int m0 = blockIdx.x * 128 + (int)(wid >> 1) * 64;
  int n0 = blockIdx.y * 128 + (int)(wid & 1) * 64;

  v8f z = {};
  v8f c[4][4];
#pragma unroll
  for (int i = 0; i < 4; i++)
#pragma unroll
    for (int j = 0; j < 4; j++) c[i][j] = z;

  for (int k0 = 0; k0 < K; k0 += 32) {
    v16h af[4], bf[4];
#pragma unroll
    for (int i = 0; i < 4; i++)
      af[i] = ld_frag(A + (size_t)(m0 + 16 * i) * K + k0, K);
#pragma unroll
    for (int j = 0; j < 4; j++)
      bf[j] = ld_frag(Bt + (size_t)(n0 + 16 * j) * K + k0, K);
#pragma unroll
    for (int i = 0; i < 4; i++)
#pragma unroll
      for (int j = 0; j < 4; j++) c[i][j] = wmma_f16(af[i], bf[j], c[i][j]);
  }

  const float sm_scale = 0.0625f;  // hd^-0.5 folded into q
  unsigned hl   = lane >> 4;
  unsigned ncol = lane & 15u;
#pragma unroll
  for (int j = 0; j < 4; j++) {
    // Each 16-wide n-tile lies in exactly one of q/k/v and one head
    // (region boundary 1280 and head width 256 are multiples of 16).
    int nt  = n0 + 16 * j;
    int reg = nt / DD;          // 0=q, 1=k, 2=v (uniform over tile)
    int nn  = nt - reg * DD;
    int h   = nn >> 8;          // head (uniform over tile)
    int ch  = (nn & 255) + (int)ncol;
    float bn = bias[nt + (int)ncol];
#pragma unroll
    for (int i = 0; i < 4; i++) {
#pragma unroll
      for (int r = 0; r < 8; r++) {
        int m  = m0 + 16 * i + r + 8 * (int)hl;
        int bb = m >> 10;   // batch
        int l  = m & 1023;  // position
        float val = c[i][j][r] + bn;
        if (reg == 0) {
          float ph = 1.0f + 0.1f * phi[l * HH + h];
          qh[(((size_t)(bb * HH + h)) * LL + l) * HD + ch] =
              (_Float16)(val * ph * sm_scale);
        } else if (reg == 1) {
          float ph = 1.0f + 0.1f * phi[l * HH + h];
          kh[(((size_t)(bb * HH + h)) * LL + l) * HD + ch] =
              (_Float16)(val * ph);
        } else {
          vt[(((size_t)(bb * HH + h)) * HD + ch) * LL + l] = (_Float16)val;
        }
      }
    }
  }
}

// ---------------- flash attention ----------------
// One wave handles one (b,h) x 16-query tile. hd=256, 32-key steps.
// Q is pre-scaled by hd^-0.5 in the QKV epilogue.
__global__ void __launch_bounds__(128) attn_kernel(
    const _Float16* __restrict__ qh, const _Float16* __restrict__ kh,
    const _Float16* __restrict__ vt, _Float16* __restrict__ oh) {
  __shared__ __align__(16) _Float16 pstage[4][16 * 32];

  unsigned wid  = threadIdx.x >> 5;
  unsigned lane = threadIdx.x & 31u;
  unsigned hl   = lane >> 4;
  unsigned ncol = lane & 15u;

  int task = blockIdx.x * 4 + (int)wid;  // 2560 tasks
  int qt = task & 63;
  int bh = task >> 6;  // b*5 + h
  int h  = bh % HH;
  int b  = bh / HH;

  const _Float16* qbase = qh + ((size_t)bh * LL + (size_t)qt * 16) * HD;
  const _Float16* kbase = kh + (size_t)bh * LL * HD;
  const _Float16* vbase = vt + (size_t)bh * HD * LL;

  v16h qf[8];
#pragma unroll
  for (int kc = 0; kc < 8; kc++) qf[kc] = ld_frag(qbase + kc * 32, HD);

  v8f z = {};
  v8f o[16];
#pragma unroll
  for (int n = 0; n < 16; n++) o[n] = z;
  float mrow[8], lrow[8];
#pragma unroll
  for (int r = 0; r < 8; r++) { mrow[r] = -3.0e38f; lrow[r] = 0.0f; }

  _Float16* P = pstage[wid];

  for (int kt = 0; kt < 32; kt++) {
    v8f s0 = z, s1 = z;
    const _Float16* kb0 = kbase + (size_t)(kt * 32) * HD;
    const _Float16* kb1 = kbase + (size_t)(kt * 32 + 16) * HD;
#pragma unroll
    for (int kc = 0; kc < 8; kc++) {
      v16h k0 = ld_frag(kb0 + kc * 32, HD);
      v16h k1 = ld_frag(kb1 + kc * 32, HD);
      s0 = wmma_f16(qf[kc], k0, s0);
      s1 = wmma_f16(qf[kc], k1, s1);
    }

    float fac[8];
#pragma unroll
    for (int r = 0; r < 8; r++) {
      float a0 = s0[r];
      float a1 = s1[r];
      float mx = fmaxf(a0, a1);
#pragma unroll
      for (int d = 8; d >= 1; d >>= 1) mx = fmaxf(mx, __shfl_xor(mx, d, 32));
      float mnew = fmaxf(mrow[r], mx);
      float f  = __expf(mrow[r] - mnew);
      float p0 = __expf(a0 - mnew);
      float p1 = __expf(a1 - mnew);
      float sum = p0 + p1;
#pragma unroll
      for (int d = 8; d >= 1; d >>= 1) sum += __shfl_xor(sum, d, 32);
      lrow[r] = lrow[r] * f + sum;
      mrow[r] = mnew;
      fac[r] = f;
      s0[r] = p0;
      s1[r] = p1;
    }

#pragma unroll
    for (int n = 0; n < 16; n++)
#pragma unroll
      for (int r = 0; r < 8; r++) o[n][r] *= fac[r];

    // Re-layout P: C-fragment -> row-major 16x32 in LDS -> A-fragment.
    // Same-wave DS ops are processed in order; no barrier needed.
#pragma unroll
    for (int r = 0; r < 8; r++) {
      int row = r + 8 * (int)hl;
      P[row * 32 + (int)ncol]      = (_Float16)s0[r];
      P[row * 32 + 16 + (int)ncol] = (_Float16)s1[r];
    }
    unsigned pm  = ncol;
    unsigned pkb = hl * 8;
    union { v16h v; v8h hh[2]; } up;
    up.hh[0] = *(const v8h*)(&P[pm * 32 + pkb]);
    up.hh[1] = *(const v8h*)(&P[pm * 32 + 16 + pkb]);
    v16h pf = up.v;

    const _Float16* vb = vbase + kt * 32;
#pragma unroll
    for (int n = 0; n < 16; n++) {
      v16h vf = ld_frag(vb + (size_t)(n * 16) * LL, LL);
      o[n] = wmma_f16(pf, vf, o[n]);
    }
  }

  float inv[8];
#pragma unroll
  for (int r = 0; r < 8; r++) inv[r] = 1.0f / lrow[r];

  int tok0 = qt * 16;
#pragma unroll
  for (int n = 0; n < 16; n++) {
#pragma unroll
    for (int r = 0; r < 8; r++) {
      int row = r + 8 * (int)hl;
      float val = o[n][r] * inv[r];
      oh[((size_t)b * LL + tok0 + row) * DD + h * HD + n * 16 + (int)ncol] =
          (_Float16)val;
    }
  }
}

// ---------------- output projection GEMM ----------------
__global__ void __launch_bounds__(128) gemm_proj(
    const _Float16* __restrict__ A, const _Float16* __restrict__ Bt,
    const float* __restrict__ bias, float* __restrict__ out) {
  const int K = DD;
  unsigned wid  = threadIdx.x >> 5;
  unsigned lane = threadIdx.x & 31u;
  int m0 = blockIdx.x * 128 + (int)(wid >> 1) * 64;
  int n0 = blockIdx.y * 128 + (int)(wid & 1) * 64;

  v8f z = {};
  v8f c[4][4];
#pragma unroll
  for (int i = 0; i < 4; i++)
#pragma unroll
    for (int j = 0; j < 4; j++) c[i][j] = z;

  for (int k0 = 0; k0 < K; k0 += 32) {
    v16h af[4], bf[4];
#pragma unroll
    for (int i = 0; i < 4; i++)
      af[i] = ld_frag(A + (size_t)(m0 + 16 * i) * K + k0, K);
#pragma unroll
    for (int j = 0; j < 4; j++)
      bf[j] = ld_frag(Bt + (size_t)(n0 + 16 * j) * K + k0, K);
#pragma unroll
    for (int i = 0; i < 4; i++)
#pragma unroll
      for (int j = 0; j < 4; j++) c[i][j] = wmma_f16(af[i], bf[j], c[i][j]);
  }

  unsigned hl   = lane >> 4;
  unsigned ncol = lane & 15u;
#pragma unroll
  for (int i = 0; i < 4; i++) {
#pragma unroll
    for (int j = 0; j < 4; j++) {
      int n = n0 + 16 * j + (int)ncol;
      float bn = bias[n];
#pragma unroll
      for (int r = 0; r < 8; r++) {
        int m = m0 + 16 * i + r + 8 * (int)hl;
        out[(size_t)m * DD + n] = c[i][j][r] + bn;
      }
    }
  }
}

// ---------------- launch ----------------

extern "C" void kernel_launch(void* const* d_in, const int* in_sizes, int n_in,
                              void* d_out, int out_size, void* d_ws,
                              size_t ws_size, hipStream_t stream) {
  const float* x     = (const float*)d_in[0];
  const float* Wqkv  = (const float*)d_in[1];
  const float* bqkv  = (const float*)d_in[2];
  const float* Wproj = (const float*)d_in[3];
  const float* bproj = (const float*)d_in[4];
  const float* phi   = (const float*)d_in[5];

  char* ws = (char*)d_ws;
  size_t off = 0;
  auto alloc = [&](size_t bytes) -> char* {
    char* p = ws + off;
    off = (off + bytes + 255) & ~(size_t)255;
    return p;
  };

  const size_t n_x   = (size_t)BB * LL * DD;       // 10,485,760
  const size_t n_qkv = (size_t)DD * N3;            // 4,915,200
  const size_t n_prj = (size_t)DD * DD;            // 1,638,400
  const size_t n_hd  = (size_t)BB * HH * LL * HD;  // 10,485,760

  _Float16* xh     = (_Float16*)alloc(n_x * 2);
  _Float16* wqkvT  = (_Float16*)alloc(n_qkv * 2);
  _Float16* wprojT = (_Float16*)alloc(n_prj * 2);
  _Float16* qh     = (_Float16*)alloc(n_hd * 2);
  _Float16* kh     = (_Float16*)alloc(n_hd * 2);
  _Float16* vt     = (_Float16*)alloc(n_hd * 2);
  _Float16* oh     = (_Float16*)alloc(n_x * 2);

  cvt_f32_to_f16<<<(int)((n_x + 255) / 256), 256, 0, stream>>>(x, xh, (int)n_x);
  transpose_cvt<<<dim3(DD / 32, N3 / 32), 256, 0, stream>>>(Wqkv, wqkvT, DD,
                                                            N3);
  transpose_cvt<<<dim3(DD / 32, DD / 32), 256, 0, stream>>>(Wproj, wprojT, DD,
                                                            DD);

  // QKV: M=8192 (64 tiles of 128), N=3840 (30 tiles of 128)
  gemm_qkv<<<dim3(64, 30), 128, 0, stream>>>(xh, wqkvT, bqkv, phi, qh, kh, vt);

  // Attention: 8*5*64 = 2560 wave-tasks, 4 waves/block
  attn_kernel<<<640, 128, 0, stream>>>(qh, kh, vt, oh);

  // Proj: M=8192, N=1280 (10 tiles of 128)
  gemm_proj<<<dim3(64, 10), 128, 0, stream>>>(oh, wprojT, bproj, (float*)d_out);
}
